// GCN_28286654612181
// MI455X (gfx1250) — matrix operand order repaired and checked by reading, not compiled, not run on hardware
//
#include <hip/hip_runtime.h>

// ---------------------------------------------------------------------------
// GCN (2x GCNConv + 3-layer MLP head + softmax) for gfx1250 / MI455X.
// GEMMs via v_wmma_f32_16x16x32_bf16 on pre-swizzled bf16 fragments.
// Graph aggregation via fp32 global atomics. All fp32 activations.
// ---------------------------------------------------------------------------

typedef __attribute__((ext_vector_type(16))) __bf16 v16bf;
typedef __attribute__((ext_vector_type(8)))  float  v8f;

static __device__ __forceinline__ unsigned short f2bf(float f) {
  unsigned int u = __float_as_uint(f);
  u += 0x7FFFu + ((u >> 16) & 1u);          // round-to-nearest-even
  return (unsigned short)(u >> 16);
}

// ---------------- degree / normalization ----------------------------------
__global__ void k_deg_init(float* deg, int n) {
  int i = blockIdx.x * blockDim.x + threadIdx.x;
  if (i < n) deg[i] = 1.0f;                 // self-loop
}

__global__ void k_deg_accum(const int* __restrict__ ei, float* deg, int E) {
  int e = blockIdx.x * blockDim.x + threadIdx.x;
  if (e < E) atomicAdd(deg + ei[(size_t)E + e], 1.0f);   // dst row of edge_index
}

__global__ void k_dinv(const float* __restrict__ deg, float* dinv, int n) {
  int i = blockIdx.x * blockDim.x + threadIdx.x;
  if (i < n) dinv[i] = rsqrtf(deg[i]);      // deg >= 1 always
}

// ---------------- operand swizzling into WMMA fragment layout -------------
// A fragment blob: tile t = mtile*KT + kt, 32 lanes x 16 halfs, lane-major.
// ISA A-layout (16x32 bf16): lane<16 -> row=lane, halfs 0-7 = K0-7, 8-15 = K16-23;
// lanes 16-31 shift K by +8.  => k = i + (i&8) + ((lane&16)>>1)
__global__ void k_convA(const float* __restrict__ X, unsigned short* __restrict__ af,
                        int Mtiles, int KT, int K) {
  size_t gid = (size_t)blockIdx.x * blockDim.x + threadIdx.x;
  size_t total = (size_t)Mtiles * KT * 512;
  if (gid >= total) return;
  int i    = (int)(gid & 15);
  int lane = (int)((gid >> 4) & 31);
  size_t tile = gid >> 9;
  int kt = (int)(tile % KT);
  int mt = (int)(tile / KT);
  int row = mt * 16 + (lane & 15);
  int k   = kt * 32 + i + (i & 8) + ((lane & 16) >> 1);
  af[gid] = f2bf(X[(size_t)row * K + k]);
}

// B fragment blob: tile t = ntile*KT + kt (k-contiguous per column strip).
// B-layout (32x16 bf16): lane = column; halfs 0-7 = K0-7, 8-15 = K16-23; lanes 16-31 shift K by +8 (matches A).
__global__ void k_swizW(const float* __restrict__ W, unsigned short* __restrict__ wf,
                        int KT, int NT, int Kreal, int Nreal) {
  size_t gid = (size_t)blockIdx.x * blockDim.x + threadIdx.x;
  size_t total = (size_t)KT * NT * 512;
  if (gid >= total) return;
  int i    = (int)(gid & 15);
  int lane = (int)((gid >> 4) & 31);
  size_t tile = gid >> 9;
  int kt = (int)(tile % KT);
  int nt = (int)(tile / KT);
  int col = nt * 16 + (lane & 15);
  int k   = kt * 32 + i + (i & 8) + ((lane & 16) >> 1);
  float f = (col < Nreal && k < Kreal) ? W[(size_t)k * Nreal + col] : 0.0f;
  wf[gid] = f2bf(f);
}

// ---------------- WMMA GEMM: out[M,Nout] = A(bf16) x B(bf16), f32 accum ---
// Block = 8 waves; each wave computes a 16-row x 64-col strip (4 WMMA tiles).
// NT is always a multiple of 4 (lin3 weight padded), so the k-loop body is
// branch-free: 10x global_load_b128 + 4x v_wmma, accumulate in place.
__global__ __launch_bounds__(256) void k_gemm_wmma(
    const unsigned short* __restrict__ afrag,
    const unsigned short* __restrict__ wfrag,
    const float* __restrict__ bias, float* __restrict__ out,
    int Mtiles, int KT, int Nout, int relu) {
  const int wave = threadIdx.x >> 5;
  const int lane = threadIdx.x & 31;
  const int mtile = blockIdx.y * 8 + wave;
  if (mtile >= Mtiles) return;              // wave-uniform: EXEC stays all-1s
  const int ntile0 = blockIdx.x * 4;

  v8f acc0 = {}, acc1 = {}, acc2 = {}, acc3 = {};
  const unsigned short* ap  = afrag + (size_t)mtile * KT * 512 + (size_t)lane * 16;
  const unsigned short* bp0 = wfrag + ((size_t)(ntile0 + 0) * KT) * 512 + (size_t)lane * 16;
  const unsigned short* bp1 = wfrag + ((size_t)(ntile0 + 1) * KT) * 512 + (size_t)lane * 16;
  const unsigned short* bp2 = wfrag + ((size_t)(ntile0 + 2) * KT) * 512 + (size_t)lane * 16;
  const unsigned short* bp3 = wfrag + ((size_t)(ntile0 + 3) * KT) * 512 + (size_t)lane * 16;

#pragma unroll 2
  for (int kt = 0; kt < KT; ++kt) {
    v16bf a  = *(const v16bf*)ap;
    v16bf b0 = *(const v16bf*)bp0;
    v16bf b1 = *(const v16bf*)bp1;
    v16bf b2 = *(const v16bf*)bp2;
    v16bf b3 = *(const v16bf*)bp3;
    __builtin_prefetch(ap + 4 * 512, 0, 0);          // global_prefetch_b8
    ap += 512; bp0 += 512; bp1 += 512; bp2 += 512; bp3 += 512;
    acc0 = __builtin_amdgcn_wmma_f32_16x16x32_bf16(false, a, false, b0, (short)0, acc0, false, false);
    acc1 = __builtin_amdgcn_wmma_f32_16x16x32_bf16(false, a, false, b1, (short)0, acc1, false, false);
    acc2 = __builtin_amdgcn_wmma_f32_16x16x32_bf16(false, a, false, b2, (short)0, acc2, false, false);
    acc3 = __builtin_amdgcn_wmma_f32_16x16x32_bf16(false, a, false, b3, (short)0, acc3, false, false);
  }

  // C/D layout: VGPR v -> row = v + 8*(lane>>4), col = lane&15
  const int colL  = lane & 15;
  const int rbase = mtile * 16 + ((lane >> 4) << 3);
  v8f accs[4] = {acc0, acc1, acc2, acc3};
#pragma unroll
  for (int nt = 0; nt < 4; ++nt) {
    const int col = (ntile0 + nt) * 16 + colL;
    if (col < Nout) {                       // only trims lin3's padded cols
      const float bv = bias ? bias[col] : 0.0f;
#pragma unroll
      for (int v = 0; v < 8; ++v) {
        float x = accs[nt][v] + bv;
        if (relu) x = fmaxf(x, 0.0f);
        out[(size_t)(rbase + v) * Nout + col] = x;
      }
    }
  }
}

// ---------------- graph aggregation ---------------------------------------
// h[i,:] = y[i,:] * dinv[i]^2   (self-loop term; also initializes h)
__global__ void k_agg_init(const float* __restrict__ y, const float* __restrict__ dinv,
                           float* __restrict__ h, int n, int F) {
  size_t gid = (size_t)blockIdx.x * blockDim.x + threadIdx.x;
  size_t total = (size_t)n * F;
  if (gid >= total) return;
  int i = (int)(gid / F);
  float d = dinv[i];
  h[gid] = y[gid] * d * d;
}

// h[dst,:] += y[src,:] * dinv[src]*dinv[dst]  (float4-vectorized atomics)
__global__ void k_agg_edges(const int* __restrict__ ei, const float* __restrict__ y,
                            const float* __restrict__ dinv, float* __restrict__ h,
                            int E, int F) {
  size_t gid = (size_t)blockIdx.x * blockDim.x + threadIdx.x;
  int chunks = F >> 2;
  size_t e = gid / chunks;
  int c = (int)(gid % chunks);
  if (e >= (size_t)E) return;
  int s = ei[e];
  int d = ei[(size_t)E + e];
  float nrm = dinv[s] * dinv[d];
  const float4 v = *(const float4*)(y + (size_t)s * F + (size_t)c * 4);
  float* hp = h + (size_t)d * F + (size_t)c * 4;
  atomicAdd(hp + 0, v.x * nrm);
  atomicAdd(hp + 1, v.y * nrm);
  atomicAdd(hp + 2, v.z * nrm);
  atomicAdd(hp + 3, v.w * nrm);
}

__global__ void k_bias_relu(float* __restrict__ h, const float* __restrict__ b,
                            int n, int F) {
  size_t gid = (size_t)blockIdx.x * blockDim.x + threadIdx.x;
  if (gid >= (size_t)n * F) return;
  int f = (int)(gid % F);
  h[gid] = fmaxf(h[gid] + b[f], 0.0f);
}

// ---------------- softmax over 40 classes ---------------------------------
__global__ void k_softmax(const float* __restrict__ in, float* __restrict__ out,
                          int n, int C) {
  int i = blockIdx.x * blockDim.x + threadIdx.x;
  if (i >= n) return;
  const float* r = in + (size_t)i * C;
  float m = r[0];
  for (int c = 1; c < C; ++c) m = fmaxf(m, r[c]);
  float s = 0.0f;
  for (int c = 0; c < C; ++c) {
    float e = __expf(r[c] - m);
    s += e;
    out[(size_t)i * C + c] = e;
  }
  float inv = 1.0f / s;
  for (int c = 0; c < C; ++c) out[(size_t)i * C + c] *= inv;
}

// ---------------------------------------------------------------------------
extern "C" void kernel_launch(void* const* d_in, const int* in_sizes, int n_in,
                              void* d_out, int out_size, void* d_ws, size_t ws_size,
                              hipStream_t stream) {
  (void)in_sizes; (void)n_in; (void)out_size; (void)ws_size;

  const int NN = 50000, EE = 800000;
  const int MT = NN / 16;                  // 3125 tiles, exact
  const int TPB = 256;

  const float* x   = (const float*)d_in[0];
  const int*   ei  = (const int*)  d_in[1];
  const float* w0  = (const float*)d_in[2];
  const float* b0  = (const float*)d_in[3];
  const float* w1  = (const float*)d_in[4];
  const float* b1  = (const float*)d_in[5];
  const float* lw1 = (const float*)d_in[6];
  const float* lb1 = (const float*)d_in[7];
  const float* lw2 = (const float*)d_in[8];
  const float* lb2 = (const float*)d_in[9];
  const float* lw3 = (const float*)d_in[10];
  const float* lb3 = (const float*)d_in[11];
  float* outp = (float*)d_out;

  // ---- workspace layout ----
  char* ws = (char*)d_ws;
  size_t off = 0;
  auto alloc = [&](size_t bytes) -> char* {
    char* p = ws + off;
    off = (off + bytes + 255) & ~(size_t)255;
    return p;
  };
  unsigned short* wf0 = (unsigned short*)alloc((size_t)4  * 16 * 512 * 2); // 128x256
  unsigned short* wf1 = (unsigned short*)alloc((size_t)8  * 16 * 512 * 2); // 256x256
  unsigned short* wf2 = (unsigned short*)alloc((size_t)8  * 32 * 512 * 2); // 256x512
  unsigned short* wf3 = (unsigned short*)alloc((size_t)16 * 64 * 512 * 2); // 512x1024
  unsigned short* wf4 = (unsigned short*)alloc((size_t)32 * 4  * 512 * 2); // 1024x64(pad)
  float* deg  = (float*)alloc((size_t)NN * 4);
  float* dinv = (float*)alloc((size_t)NN * 4);
  unsigned short* afrag = (unsigned short*)alloc((size_t)MT * 32 * 512 * 2); // max K=1024
  float* ybuf = (float*)alloc((size_t)NN * 256 * 4);   // pre-agg / logits
  float* pbuf = (float*)alloc((size_t)NN * 1024 * 4);
  float* qbuf = (float*)alloc((size_t)NN * 1024 * 4);

  auto cdiv = [](size_t a, size_t b) { return (unsigned)((a + b - 1) / b); };

  // ---- normalization coefficients ----
  k_deg_init <<<cdiv(NN, TPB), TPB, 0, stream>>>(deg, NN);
  k_deg_accum<<<cdiv(EE, TPB), TPB, 0, stream>>>(ei, deg, EE);
  k_dinv     <<<cdiv(NN, TPB), TPB, 0, stream>>>(deg, dinv, NN);

  // ---- weight swizzle (bf16 WMMA-B fragments; lin3 padded to NT=4) ----
  k_swizW<<<cdiv((size_t)4  * 16 * 512, TPB), TPB, 0, stream>>>(w0,  wf0, 4,  16, 128,  256);
  k_swizW<<<cdiv((size_t)8  * 16 * 512, TPB), TPB, 0, stream>>>(w1,  wf1, 8,  16, 256,  256);
  k_swizW<<<cdiv((size_t)8  * 32 * 512, TPB), TPB, 0, stream>>>(lw1, wf2, 8,  32, 256,  512);
  k_swizW<<<cdiv((size_t)16 * 64 * 512, TPB), TPB, 0, stream>>>(lw2, wf3, 16, 64, 512, 1024);
  k_swizW<<<cdiv((size_t)32 * 4  * 512, TPB), TPB, 0, stream>>>(lw3, wf4, 32, 4, 1024,   40);

  const unsigned gy = cdiv(MT, 8);
  auto gemm = [&](const unsigned short* af, const unsigned short* wf,
                  const float* bias, float* o, int KT, int NT, int Nout, int relu) {
    dim3 grid((unsigned)(NT / 4), gy);     // NT always a multiple of 4
    k_gemm_wmma<<<grid, TPB, 0, stream>>>(af, wf, bias, o, MT, KT, Nout, relu);
  };
  auto agg = [&](const float* y, float* h, const float* bias) {
    k_agg_init <<<cdiv((size_t)NN * 256, TPB), TPB, 0, stream>>>(y, dinv, h, NN, 256);
    k_agg_edges<<<cdiv((size_t)EE * 64,  TPB), TPB, 0, stream>>>(ei, y, dinv, h, EE, 256);
    k_bias_relu<<<cdiv((size_t)NN * 256, TPB), TPB, 0, stream>>>(h, bias, NN, 256);
  };

  // conv0: y = x @ W0 ; h0 = relu(agg(y) + b0) -> pbuf
  k_convA<<<cdiv((size_t)MT * 4 * 512, TPB), TPB, 0, stream>>>(x, afrag, MT, 4, 128);
  gemm(afrag, wf0, nullptr, ybuf, 4, 16, 256, 0);
  agg(ybuf, pbuf, b0);

  // conv1: y = h0 @ W1 ; h1 = relu(agg(y) + b1) -> qbuf
  k_convA<<<cdiv((size_t)MT * 8 * 512, TPB), TPB, 0, stream>>>(pbuf, afrag, MT, 8, 256);
  gemm(afrag, wf1, nullptr, ybuf, 8, 16, 256, 0);
  agg(ybuf, qbuf, b1);

  // lin1: h2 = relu(h1 @ lw1 + lb1) -> pbuf [50000,512]
  k_convA<<<cdiv((size_t)MT * 8 * 512, TPB), TPB, 0, stream>>>(qbuf, afrag, MT, 8, 256);
  gemm(afrag, wf2, lb1, pbuf, 8, 32, 512, 1);

  // lin2: h3 = relu(h2 @ lw2 + lb2) -> qbuf [50000,1024]
  k_convA<<<cdiv((size_t)MT * 16 * 512, TPB), TPB, 0, stream>>>(pbuf, afrag, MT, 16, 512);
  gemm(afrag, wf3, lb2, qbuf, 16, 64, 1024, 1);

  // lin3: logits = h3 @ lw3 + lb3 -> ybuf [50000,40]  (NT padded to 4)
  k_convA<<<cdiv((size_t)MT * 32 * 512, TPB), TPB, 0, stream>>>(qbuf, afrag, MT, 32, 1024);
  gemm(afrag, wf4, lb3, ybuf, 32, 4, 40, 0);

  // softmax
  k_softmax<<<cdiv(NN, TPB), TPB, 0, stream>>>(ybuf, outp, NN, 40);
}
